// TemporalEncoder_80195629351469
// MI455X (gfx1250) — matrix operand order repaired and checked by reading, not compiled, not run on hardware
//
#include <hip/hip_runtime.h>
#include <hip/hip_bf16.h>

typedef __attribute__((ext_vector_type(16))) _Float16 v16h;
typedef __attribute__((ext_vector_type(8)))  float    v8f;
typedef __attribute__((ext_vector_type(4)))  float    v4f;

#define B_SZ   512
#define S_SZ   2048
#define I_SZ   64
#define H_SZ   128
#define G_SZ   24      // 3 * latent_B(8) real gate columns
#define GPAD   32      // padded gate columns: two 16-wide WMMA N tiles
#define M_ROWS (B_SZ * S_SZ)

// workspace layout (bytes)
#define WC_OFF    0            // 32*64 f16  = 4096 B  (combined weight, f16)
#define BC_OFF    4096         // 32 f32     = 128 B   (combined bias)
#define GATES_OFF 4352         // M_ROWS*32 f32 = 128 MB (fits in 192MB L2)

// ---------------------------------------------------------------------------
// Kernel 1: fold the two linear layers:  Wc = W_ih @ W_in  (24x64, stored f16,
// padded to 32 rows of zeros), bc = W_ih @ b_in + b_ih (padded to 32).
// ---------------------------------------------------------------------------
__global__ void __launch_bounds__(256)
prep_kernel(const float* __restrict__ W_in, const float* __restrict__ b_in,
            const float* __restrict__ W_ih, const float* __restrict__ b_ih,
            _Float16* __restrict__ wc, float* __restrict__ bc) {
    int idx = blockIdx.x * blockDim.x + threadIdx.x;
    if (idx < GPAD * I_SZ) {
        int g = idx / I_SZ, i = idx % I_SZ;
        float acc = 0.0f;
        if (g < G_SZ) {
            for (int h = 0; h < H_SZ; ++h)
                acc = fmaf(W_ih[g * H_SZ + h], W_in[h * I_SZ + i], acc);
        }
        wc[g * I_SZ + i] = (_Float16)acc;
    } else if (idx < GPAD * I_SZ + GPAD) {
        int g = idx - GPAD * I_SZ;
        float acc = 0.0f;
        if (g < G_SZ) {
            for (int h = 0; h < H_SZ; ++h)
                acc = fmaf(W_ih[g * H_SZ + h], b_in[h], acc);
            acc += b_ih[g];
        }
        bc[g] = acc;
    }
}

// ---------------------------------------------------------------------------
// Kernel 2: gates[m, 0:32] = A[m, 0:64] @ Wc^T (+ bias seeded into C operand)
// via WMMA f16->f32.  One wave per 16-row tile; TILES_PER_WAVE tiles per wave
// with hoisted B operands and hoisted bias-C seeds.
// A-operand layout (16-bit A 16x32, wave32): lane<16 row M=lane holds
// K 0..15; lane>=16 holds K 16..31 (code's contiguous order, same as B). B-operand (32x16):
// lane<16 col N=lane holds K 0..15; lane>=16 holds K 16..31.
// D layout: lane holds col N=(lane&15); VGPR v = row (v + 8*(lane>=16)).
// All stores unconditional: a 16-row tile writes a dense 2KB region.
// ---------------------------------------------------------------------------
#define WAVES_PER_BLOCK 8
#define TILES_PER_WAVE  8

__global__ void __launch_bounds__(256)
gates_gemm_kernel(const float* __restrict__ A, const _Float16* __restrict__ wc,
                  const float* __restrict__ bc, float* __restrict__ gates) {
    const int lane = threadIdx.x & 31;
    const int wave = threadIdx.x >> 5;
    const int hi   = lane >> 4;     // 0: lanes 0-15, 1: lanes 16-31
    const int nlo  = lane & 15;

    // Loop-invariant B operands: bmat[ntile][khalf]
    v16h bmat[2][2];
#pragma unroll
    for (int nt = 0; nt < 2; ++nt) {
        int g = nt * 16 + nlo;
#pragma unroll
        for (int kh = 0; kh < 2; ++kh) {
            const _Float16* src = wc + g * I_SZ + kh * 32 + hi * 16; // 32B aligned
            bmat[nt][kh] = *(const v16h*)src;
        }
    }
    // Bias folded into the accumulator seed (per-column constant across rows).
    const float bias0 = bc[nlo];
    const float bias1 = bc[16 + nlo];
    v8f cseed0, cseed1;
#pragma unroll
    for (int e = 0; e < 8; ++e) { cseed0[e] = bias0; cseed1[e] = bias1; }

    const long tile0 = ((long)blockIdx.x * WAVES_PER_BLOCK + wave) * TILES_PER_WAVE;
#pragma unroll 2
    for (int tt = 0; tt < TILES_PER_WAVE; ++tt) {
        const long m0 = (tile0 + tt) * 16;
        const float* arow = A + (m0 + nlo) * (long)I_SZ;

        v16h amat[2];
#pragma unroll
        for (int kh = 0; kh < 2; ++kh) {
            const int base = kh * 32 + hi * 16;
            v4f c0 = *(const v4f*)(arow + base);
            v4f c1 = *(const v4f*)(arow + base + 4);
            v4f c2 = *(const v4f*)(arow + base + 8);
            v4f c3 = *(const v4f*)(arow + base + 12);
            v16h a;
#pragma unroll
            for (int e = 0; e < 4; ++e) {
                a[e]      = (_Float16)c0[e];
                a[4 + e]  = (_Float16)c1[e];
                a[8 + e]  = (_Float16)c2[e];
                a[12 + e] = (_Float16)c3[e];
            }
            amat[kh] = a;
        }

        // N-tile 0: gate cols 0..15 (xr | xz), C seeded with bias
        v8f c = __builtin_amdgcn_wmma_f32_16x16x32_f16(false, amat[0], false, bmat[0][0],
                                                       (short)0, cseed0, false, false);
        c = __builtin_amdgcn_wmma_f32_16x16x32_f16(false, amat[1], false, bmat[0][1],
                                                   (short)0, c, false, false);
        // N-tile 1: gate cols 16..31 (xn + 8 padding cols)
        v8f d = __builtin_amdgcn_wmma_f32_16x16x32_f16(false, amat[0], false, bmat[1][0],
                                                       (short)0, cseed1, false, false);
        d = __builtin_amdgcn_wmma_f32_16x16x32_f16(false, amat[1], false, bmat[1][1],
                                                   (short)0, d, false, false);

        // Unconditional stores into padded 32-col layout: tile = dense 2KB.
        float* grow = gates + m0 * GPAD + nlo;
#pragma unroll
        for (int v = 0; v < 8; ++v) {
            const long roff = (long)(v + 8 * hi) * GPAD;
            grow[roff]      = c[v];
            grow[roff + 16] = d[v];
        }
    }
}

// ---------------------------------------------------------------------------
// Kernel 3: GRU scan. 8 lanes per batch element (lane j = hidden unit j).
// 8x8 W_hh matvecs via __shfl(width=8) broadcast of h. Gates stream from L2.
// ---------------------------------------------------------------------------
__global__ void __launch_bounds__(256)
gru_scan_kernel(const float* __restrict__ gates, const float* __restrict__ W_hh,
                const float* __restrict__ b_hh, float* __restrict__ out) {
    const int tid = blockIdx.x * blockDim.x + threadIdx.x;
    const int b = tid >> 3;
    const int j = tid & 7;

    float wr[8], wz[8], wn[8];
#pragma unroll
    for (int k = 0; k < 8; ++k) {
        wr[k] = W_hh[j * 8 + k];          // Whr row j
        wz[k] = W_hh[(8 + j) * 8 + k];    // Whz row j
        wn[k] = W_hh[(16 + j) * 8 + k];   // Whn row j
    }
    const float bhr = b_hh[j], bhz = b_hh[8 + j], bhn = b_hh[16 + j];

    float h = 0.0f;
    const float* gp = gates + (long)b * S_SZ * GPAD;

    for (int t = 0; t < S_SZ; ++t) {
        __builtin_prefetch(gp + 16 * GPAD, 0, 0);   // global_prefetch, ~16 steps ahead
        const float xr = gp[j];
        const float xz = gp[8 + j];
        const float xn = gp[16 + j];
        gp += GPAD;

        float sr = 0.0f, sz = 0.0f, sn = 0.0f;
#pragma unroll
        for (int k = 0; k < 8; ++k) {
            const float hk = __shfl(h, k, 8);
            sr = fmaf(wr[k], hk, sr);
            sz = fmaf(wz[k], hk, sz);
            sn = fmaf(wn[k], hk, sn);
        }
        const float r  = 1.0f / (1.0f + __expf(-(xr + sr + bhr)));
        const float z  = 1.0f / (1.0f + __expf(-(xz + sz + bhz)));
        const float a  = xn + r * (sn + bhn);
        const float e2 = __expf(-2.0f * a);
        const float n  = (1.0f - e2) / (1.0f + e2);
        h = (1.0f - z) * n + z * h;
    }
    out[b * 8 + j] = h;   // (512, 8) row-major
}

// ---------------------------------------------------------------------------
extern "C" void kernel_launch(void* const* d_in, const int* in_sizes, int n_in,
                              void* d_out, int out_size, void* d_ws, size_t ws_size,
                              hipStream_t stream) {
    const float* A    = (const float*)d_in[0];  // (512, 2048, 64)
    const float* W_in = (const float*)d_in[1];  // (128, 64)
    const float* b_in = (const float*)d_in[2];  // (128,)
    const float* W_ih = (const float*)d_in[3];  // (24, 128)
    const float* W_hh = (const float*)d_in[4];  // (24, 8)
    const float* b_ih = (const float*)d_in[5];  // (24,)
    const float* b_hh = (const float*)d_in[6];  // (24,)
    float* out = (float*)d_out;                 // (512, 8)

    char* ws = (char*)d_ws;
    _Float16* wc    = (_Float16*)(ws + WC_OFF);
    float*    bc    = (float*)(ws + BC_OFF);
    float*    gates = (float*)(ws + GATES_OFF); // 512*2048*32 f32 = 128 MB

    // 1) fold linear layers into 24x64 weight + bias
    prep_kernel<<<(GPAD * I_SZ + GPAD + 255) / 256, 256, 0, stream>>>(
        W_in, b_in, W_ih, b_ih, wc, bc);

    // 2) WMMA GEMM: 65536 sixteen-row tiles / (8 waves * 8 tiles) = 1024 blocks
    const int n_tiles = M_ROWS / 16;
    const int blocks  = n_tiles / (WAVES_PER_BLOCK * TILES_PER_WAVE);
    gates_gemm_kernel<<<blocks, 32 * WAVES_PER_BLOCK, 0, stream>>>(A, wc, bc, gates);

    // 3) GRU scan: 512 batches * 8 lanes = 4096 threads
    gru_scan_kernel<<<(B_SZ * 8) / 256, 256, 0, stream>>>(gates, W_hh, b_hh, out);
}
